// S4BaseBlock_40673340293926
// MI455X (gfx1250) — compile-verified, matrix operand order checked by reading, not run on hardware
//
#include <hip/hip_runtime.h>

// ---------------------------------------------------------------------------
// S4 block for MI455X (gfx1250, wave32, WMMA + TDM):
//   y = causal_conv(x, K),  K[i] = C A_bar^i B_bar, via exact chunked SSM:
//     s_{c+1} = P s_c + U x_c      (P = A_bar^T, chunk T = 256)
//     y_c     = M x_c + W s_c      (M = Toeplitz(K), W[tau] = C A_bar^{tau+1})
//   GEMMs in bf16 WMMA (f32 accumulate). Operands stored in LDS pre-swizzled
//   into the WMMA register image -> every operand fetch is 2x ds_load_b128.
//   Table/state LDS fills use the Tensor Data Mover (tensor_load_to_lds).
// ---------------------------------------------------------------------------

typedef __bf16 bf16;
typedef __attribute__((ext_vector_type(16))) __bf16 v16bf;
typedef __attribute__((ext_vector_type(8)))  float  v8f;
typedef unsigned int u32x4 __attribute__((ext_vector_type(4)));
typedef int          i32x4 __attribute__((ext_vector_type(4)));
typedef int          i32x8 __attribute__((ext_vector_type(8)));

#define NDIM 256
#define TCH  256
#define NCH  64
#define LSEQ 16384
#define BSZ  512
#define DT_H 0.05f

#if defined(__has_builtin)
#if __has_builtin(__builtin_amdgcn_tensor_load_to_lds) && \
    __has_builtin(__builtin_amdgcn_s_wait_tensorcnt)
#define USE_TDM 1
#endif
#endif
#ifndef USE_TDM
#define USE_TDM 0
#endif

__device__ __forceinline__ v8f wmma_bf16(v16bf a, v16bf b, v8f c) {
  return __builtin_amdgcn_wmma_f32_16x16x32_bf16(false, a, false, b,
                                                 (short)0, c, false, false);
}

// Swizzle: element (mn in [0,16), k in [0,32)) of a 16x32 A-tile / 32x16 B-tile
//   lane = mn + 16*((k>>3)&1) ; elem = (k&7) | ((k>>4)<<3)   (ISA 7.12.2)
// Tiles stored as [lane][16 bf16 contiguous] = 512 bf16 = 1 KB.
__device__ __forceinline__ int swz_off(int mn, int k) {
  int lane = mn + (((k >> 3) & 1) << 4);
  int e    = (k & 7) | ((k >> 4) << 3);
  return lane * 16 + e;
}

// Load one pre-swizzled tile: 32 bytes/lane, contiguous -> 2x ds_load_b128.
__device__ __forceinline__ v16bf load_tile_swz(const bf16* p) {
  int lane = threadIdx.x & 31;
  union { uint4 u[2]; v16bf v; } t;
  const uint4* q = (const uint4*)p;
  t.u[0] = q[lane * 2];
  t.u[1] = q[lane * 2 + 1];
  return t.v;
}

#if USE_TDM
// Tensor Data Mover: 2-D tile load global->LDS, data_size = 2 bytes.
// D# packing per CDNA5 ISA 8.3-8.5 (group0: count/lds/global/type,
// group1: data_size, dims, tile dims, dim0 stride). Groups 2/3 zero (2-D).
__device__ __forceinline__ void tdm_load_2d(unsigned lds_off, const void* gaddr,
                                            unsigned dim0, unsigned dim1,
                                            unsigned tile0, unsigned tile1,
                                            unsigned stride0_elems) {
  unsigned long long ga = (unsigned long long)gaddr;
  u32x4 g0;
  g0.x = 1u;                                         // count=1, user D#
  g0.y = lds_off;                                    // lds_addr (bytes)
  g0.z = (unsigned)(ga & 0xffffffffu);               // global_addr[31:0]
  g0.w = (unsigned)((ga >> 32) & 0x1ffffffu) | (2u << 30);  // addr hi | type=2
  i32x8 g1;
  g1[0] = (int)(1u << 16);                           // data_size=1 (2B), mask=0
  g1[1] = (int)(dim0 << 16);                         // tensor_dim0[15:0]
  g1[2] = (int)((dim0 >> 16) | (dim1 << 16));        // dim0 hi | dim1 lo
  g1[3] = (int)((dim1 >> 16) | (tile0 << 16));       // dim1 hi | tile_dim0
  g1[4] = (int)(tile1 & 0xffffu);                    // tile_dim1 (tile_dim2=0)
  g1[5] = (int)stride0_elems;                        // tensor_dim0_stride lo
  g1[6] = 0;                                         // stride hi | dim1_stride
  g1[7] = 0;
  i32x4 z4 = {0, 0, 0, 0};
#if __clang_major__ >= 23
  i32x8 z8 = {0, 0, 0, 0, 0, 0, 0, 0};
  __builtin_amdgcn_tensor_load_to_lds(g0, g1, z4, z4, z8, 0);
#else
  __builtin_amdgcn_tensor_load_to_lds(g0, g1, z4, z4, 0);
#endif
}
#endif

// ---------------------------------------------------------------------------
// Precompute kernels (O(N^3) ~ 17M ops, negligible)
// ---------------------------------------------------------------------------
__global__ __launch_bounds__(1024)
void gj_invert_k(const float* __restrict__ A, float* __restrict__ G) {
  __shared__ float fac[NDIM];
  const int tid = threadIdx.x;
  for (int i = tid; i < NDIM * 512; i += 1024) {
    int r = i >> 9, c = i & 511;
    float v;
    if (c < NDIM) v = (r == c ? 1.f : 0.f) - DT_H * A[r * NDIM + c];
    else          v = ((c - NDIM) == r) ? 1.f : 0.f;
    G[i] = v;
  }
  __syncthreads();
  for (int p = 0; p < NDIM; ++p) {
    float piv = G[p * 512 + p];
    __syncthreads();
    float inv = 1.0f / piv;
    for (int j = tid; j < 512; j += 1024) G[p * 512 + j] *= inv;
    __syncthreads();
    for (int r = tid; r < NDIM; r += 1024) fac[r] = G[r * 512 + p];
    __syncthreads();
    for (int i = tid; i < NDIM * 512; i += 1024) {
      int r = i >> 9, j = i & 511;
      if (r != p) G[i] -= fac[r] * G[p * 512 + j];
    }
    __syncthreads();
  }
}

__global__ __launch_bounds__(256)
void abar_k(const float* __restrict__ G, const float* __restrict__ A,
            float* __restrict__ Abar) {
  __shared__ float row[NDIM];
  int r = blockIdx.x, c = threadIdx.x;
  row[c] = G[r * 512 + 256 + c];
  __syncthreads();
  float s = 0.f;
  for (int k = 0; k < NDIM; ++k) s += row[k] * A[k * NDIM + c];
  Abar[r * NDIM + c] = row[c] + DT_H * s;
}

__global__ __launch_bounds__(256)
void bbar_k(const float* __restrict__ G, const float* __restrict__ Bv,
            float* __restrict__ Bbar) {
  int n = threadIdx.x;
  float s = 0.f;
  for (int k = 0; k < NDIM; ++k) s += G[n * 512 + 256 + k] * Bv[k];
  Bbar[n] = 0.1f * s;
}

__global__ __launch_bounds__(256)
void chains_k(const float* __restrict__ Abar, const float* __restrict__ Bbar,
              const float* __restrict__ Cv, float* __restrict__ K,
              float* __restrict__ Uf, float* __restrict__ Wf) {
  __shared__ float v[NDIM], vn[NDIM], red[NDIM];
  int n = threadIdx.x;
  v[n] = Bbar[n];
  __syncthreads();
  for (int i = 0; i < TCH; ++i) {
    red[n] = Cv[n] * v[n];
    Uf[n * TCH + (TCH - 1 - i)] = v[n];
    __syncthreads();
    for (int s = 128; s > 0; s >>= 1) {
      if (n < s) red[n] += red[n + s];
      __syncthreads();
    }
    if (n == 0) K[i] = red[0];
    float s2 = 0.f;
    for (int k = 0; k < NDIM; ++k) s2 += Abar[n * NDIM + k] * v[k];
    vn[n] = s2;
    __syncthreads();
    v[n] = vn[n];
    __syncthreads();
  }
  float s0 = 0.f;
  for (int k = 0; k < NDIM; ++k) s0 += Cv[k] * Abar[k * NDIM + n];
  v[n] = s0;
  __syncthreads();
  for (int t = 0; t < TCH; ++t) {
    Wf[t * NDIM + n] = v[n];
    float s2 = 0.f;
    for (int k = 0; k < NDIM; ++k) s2 += v[k] * Abar[k * NDIM + n];
    __syncthreads();
    v[n] = s2;
    __syncthreads();
  }
}

__global__ __launch_bounds__(256)
void square_k(const float* __restrict__ src, float* __restrict__ dst) {
  __shared__ float row[NDIM];
  int r = blockIdx.x, c = threadIdx.x;
  row[c] = src[r * NDIM + c];
  __syncthreads();
  float s = 0.f;
  for (int k = 0; k < NDIM; ++k) s += row[k] * src[k * NDIM + c];
  dst[r * NDIM + c] = s;
}

// f32 tables -> bf16, pre-swizzled into A-operand tile layout:
//   tile = (m>>4)*8 + (k>>5); inside: swz_off(m&15, k&31)
__global__ __launch_bounds__(256)
void tables_k(const float* __restrict__ P, const float* __restrict__ Uf,
              const float* __restrict__ Wf, const float* __restrict__ K,
              bf16* __restrict__ Pb, bf16* __restrict__ Ub,
              bf16* __restrict__ Wb, bf16* __restrict__ Mb) {
  int i = blockIdx.x * 256 + threadIdx.x;   // 0..65535, i = m*256 + k
  int m = i >> 8, k = i & 255;
  int dst = ((m >> 4) * 8 + (k >> 5)) * 512 + swz_off(m & 15, k & 31);
  int which = blockIdx.y;
  if (which == 0)      Pb[dst] = (bf16)P[i];
  else if (which == 1) Ub[dst] = (bf16)Uf[i];
  else if (which == 2) Wb[dst] = (bf16)Wf[i];
  else                 Mb[dst] = (m >= k) ? (bf16)K[m - k] : (bf16)0.f;
}

// ---------------------------------------------------------------------------
// Phase 1: S_{c+1} = P S_c + U X_c ; 32 WGs x 16 batches; P,U LDS-resident.
// states stored tile-swizzled: [(c*8+kk)*32 + (b>>4)]*512 + swz_off(b&15,n&31)
// ---------------------------------------------------------------------------
__global__ __launch_bounds__(256)
void phase1_k(const float* __restrict__ x, const bf16* __restrict__ Pg,
              const bf16* __restrict__ Ug, bf16* __restrict__ states) {
  extern __shared__ char smem[];
  bf16* sP = (bf16*)smem;                 // 65536 (swizzled tiles)
  bf16* sU = sP + NDIM * NDIM;            // 65536
  bf16* sS = sU + NDIM * NDIM;            // 8 tiles * 512
  bf16* sX = sS + 8 * 512;                // 8 tiles * 512

  const int tid  = threadIdx.x;
  const int wave = tid >> 5;
  const int lane = tid & 31;
  const int b0   = blockIdx.x * 16;
  const int g    = b0 >> 4;

#if USE_TDM
  if (wave == 0) {
    // contiguous 128 KB each, expressed as 512 x 128 elements
    tdm_load_2d((unsigned)(size_t)(void*)sP, Pg, 512u, 128u, 512u, 128u, 512u);
    tdm_load_2d((unsigned)(size_t)(void*)sU, Ug, 512u, 128u, 512u, 128u, 512u);
    __builtin_amdgcn_s_wait_tensorcnt(0);
  }
#else
  {
    const uint4* gP = (const uint4*)Pg;
    const uint4* gU = (const uint4*)Ug;
    uint4* lP = (uint4*)sP;
    uint4* lU = (uint4*)sU;
    for (int i = tid; i < NDIM * NDIM / 8; i += 256) { lP[i] = gP[i]; lU[i] = gU[i]; }
  }
#endif

  // S_0 = 0 (LDS + global, swizzle-invariant)
  for (int i = tid; i < 8 * 512; i += 256) {
    sS[i] = (bf16)0.f;
    int kk = i >> 9, j = i & 511;
    states[(size_t)(kk * 32 + g) * 512 + j] = (bf16)0.f;
  }
  __syncthreads();

  const int r0  = wave * 32;
  const int nn  = lane & 15;
  const int mb  = (lane >> 4) << 3;
  const int t0  = (r0 >> 4) * 8;          // A-tile row index base (x8 kk)
  const int t1  = t0 + 8;

  for (int c = 0; c < NCH - 1; ++c) {
    // X_c -> LDS, B-operand swizzle (k index = tid, col = bb)
    const int kk = tid >> 5, so = swz_off(0, tid & 31);   // col added below
#pragma unroll 4
    for (int bb = 0; bb < 16; ++bb) {
      float v = x[(size_t)(b0 + bb) * LSEQ + c * TCH + tid];
      sX[kk * 512 + so + bb * 16] = (bf16)v;   // lane = bb + hi-bit of k
    }
    __syncthreads();

    v8f acc0 = {}, acc1 = {};
#pragma unroll
    for (int q = 0; q < 8; ++q) {
      v16bf bS  = load_tile_swz(sS + q * 512);
      v16bf bX  = load_tile_swz(sX + q * 512);
      v16bf aP0 = load_tile_swz(sP + (t0 + q) * 512);
      v16bf aP1 = load_tile_swz(sP + (t1 + q) * 512);
      v16bf aU0 = load_tile_swz(sU + (t0 + q) * 512);
      v16bf aU1 = load_tile_swz(sU + (t1 + q) * 512);
      acc0 = wmma_bf16(aP0, bS, acc0);
      acc0 = wmma_bf16(aU0, bX, acc0);
      acc1 = wmma_bf16(aP1, bS, acc1);
      acc1 = wmma_bf16(aU1, bX, acc1);
    }
    __syncthreads();   // all reads of sS done before overwrite

    bf16* gs = states + (size_t)(c + 1) * 8 * 32 * 512 + (size_t)g * 512;
#pragma unroll
    for (int r = 0; r < 8; ++r) {
      int m0 = r0 + r + mb;
      int m1 = m0 + 16;
      bf16 h0 = (bf16)acc0[r];
      bf16 h1 = (bf16)acc1[r];
      int o0 = (m0 >> 5) * 512 + swz_off(nn, m0 & 31);
      int o1 = (m1 >> 5) * 512 + swz_off(nn, m1 & 31);
      sS[o0] = h0;
      sS[o1] = h1;
      gs[(size_t)(m0 >> 5) * (32 * 512) + swz_off(nn, m0 & 31)] = h0;
      gs[(size_t)(m1 >> 5) * (32 * 512) + swz_off(nn, m1 & 31)] = h1;
    }
    __syncthreads();
  }
}

// ---------------------------------------------------------------------------
// Phase 2: Y_c = M X_c + W S_c ; grid 64 chunks x 16 batch-tiles (32 batches).
// ---------------------------------------------------------------------------
__global__ __launch_bounds__(256)
void phase2_k(const float* __restrict__ x, const bf16* __restrict__ states,
              const bf16* __restrict__ Mg, const bf16* __restrict__ Wg,
              float* __restrict__ y) {
  extern __shared__ char smem[];
  bf16* sM = (bf16*)smem;                 // 65536 (swizzled A tiles)
  bf16* sW = sM + NDIM * NDIM;            // 65536
  bf16* sX = sW + NDIM * NDIM;            // 16 tiles * 512
  bf16* sS = sX + 16 * 512;               // 16 tiles * 512

  const int tid  = threadIdx.x;
  const int wave = tid >> 5;
  const int lane = tid & 31;
  const int c    = blockIdx.x;
  const int b0   = blockIdx.y * 32;
  const int g0   = b0 >> 4;

#if USE_TDM
  if (wave == 0) {
    tdm_load_2d((unsigned)(size_t)(void*)sM, Mg, 512u, 128u, 512u, 128u, 512u);
    tdm_load_2d((unsigned)(size_t)(void*)sW, Wg, 512u, 128u, 512u, 128u, 512u);
    // states: 8 rows (kk) of 1024 contiguous elems (2 col-groups), stride 32*512
    tdm_load_2d((unsigned)(size_t)(void*)sS,
                states + (size_t)c * 8 * 32 * 512 + (size_t)g0 * 512,
                1024u, 8u, 1024u, 8u, 32u * 512u);
    __builtin_amdgcn_s_wait_tensorcnt(0);
  }
#else
  {
    const uint4* gM = (const uint4*)Mg;
    const uint4* gW = (const uint4*)Wg;
    uint4* lM = (uint4*)sM;
    uint4* lW = (uint4*)sW;
    for (int i = tid; i < NDIM * NDIM / 8; i += 256) { lM[i] = gM[i]; lW[i] = gW[i]; }
    const uint4* gS = (const uint4*)(states + (size_t)c * 8 * 32 * 512);
    uint4* lS = (uint4*)sS;
    for (int i = tid; i < 16 * 512 / 8; i += 256) {
      int t = i >> 6, j = i & 63;                 // tile (kk*2+cg), uint4 index
      int kk = t >> 1, cg = t & 1;
      lS[i] = gS[(size_t)(kk * 32 + g0 + cg) * 64 + j];
    }
  }
#endif

  // X_c -> LDS, B-operand swizzle; tile = kk*2 + (bb>>4)
  {
    const int kk = tid >> 5, so = swz_off(0, tid & 31);
#pragma unroll 4
    for (int bb = 0; bb < 32; ++bb) {
      float v = x[(size_t)(b0 + bb) * LSEQ + c * TCH + tid];
      sX[(kk * 2 + (bb >> 4)) * 512 + so + (bb & 15) * 16] = (bf16)v;
    }
  }
  __syncthreads();

  const int r0 = wave * 32;
  const int t0 = (r0 >> 4) * 8;
  const int t1 = t0 + 8;
  v8f acc[4] = {{}, {}, {}, {}};
#pragma unroll
  for (int q = 0; q < 8; ++q) {
    v16bf bX0 = load_tile_swz(sX + (q * 2 + 0) * 512);
    v16bf bX1 = load_tile_swz(sX + (q * 2 + 1) * 512);
    v16bf bS0 = load_tile_swz(sS + (q * 2 + 0) * 512);
    v16bf bS1 = load_tile_swz(sS + (q * 2 + 1) * 512);
    v16bf aM0 = load_tile_swz(sM + (t0 + q) * 512);
    v16bf aM1 = load_tile_swz(sM + (t1 + q) * 512);
    v16bf aW0 = load_tile_swz(sW + (t0 + q) * 512);
    v16bf aW1 = load_tile_swz(sW + (t1 + q) * 512);
    acc[0] = wmma_bf16(aM0, bX0, acc[0]);  acc[0] = wmma_bf16(aW0, bS0, acc[0]);
    acc[1] = wmma_bf16(aM0, bX1, acc[1]);  acc[1] = wmma_bf16(aW0, bS1, acc[1]);
    acc[2] = wmma_bf16(aM1, bX0, acc[2]);  acc[2] = wmma_bf16(aW1, bS0, acc[2]);
    acc[3] = wmma_bf16(aM1, bX1, acc[3]);  acc[3] = wmma_bf16(aW1, bS1, acc[3]);
  }
  __syncthreads();   // done with sM; reuse as f32 Y staging (128 KB region)

  float* sY = (float*)smem;               // 256 x 32 f32 = 32 KB
  const int nn  = lane & 15;
  const int mbv = (lane >> 4) << 3;
#pragma unroll
  for (int rt = 0; rt < 2; ++rt)
#pragma unroll
    for (int ct = 0; ct < 2; ++ct)
#pragma unroll
      for (int r = 0; r < 8; ++r) {
        int tau = r0 + 16 * rt + r + mbv;
        int col = 16 * ct + nn;
        sY[tau * 32 + col] = acc[rt * 2 + ct][r];
      }
  __syncthreads();

  for (int i = tid; i < TCH * 32; i += 256) {
    int bb = i >> 8, t = i & 255;
    y[(size_t)(b0 + bb) * LSEQ + c * TCH + t] = sY[t * 32 + bb];
  }
}

// ---------------------------------------------------------------------------
// Host orchestration (all launches on `stream`, graph-capture safe).
// ---------------------------------------------------------------------------
extern "C" void kernel_launch(void* const* d_in, const int* in_sizes, int n_in,
                              void* d_out, int out_size, void* d_ws, size_t ws_size,
                              hipStream_t stream) {
  const float* x  = (const float*)d_in[0];
  const float* A  = (const float*)d_in[1];
  const float* Bv = (const float*)d_in[2];
  const float* Cv = (const float*)d_in[3];
  float* y = (float*)d_out;

  char* ws = (char*)d_ws;
  size_t off = 0;
  auto alloc = [&](size_t bytes) {
    void* p = ws + off;
    off += (bytes + 255) & ~(size_t)255;
    return p;
  };
  float* G     = (float*)alloc(256 * 512 * 4);
  float* Abar  = (float*)alloc(256 * 256 * 4);
  float* Bbar  = (float*)alloc(256 * 4);
  float* Kk    = (float*)alloc(256 * 4);
  float* Uf    = (float*)alloc(256 * 256 * 4);
  float* Wf    = (float*)alloc(256 * 256 * 4);
  float* pow0  = (float*)alloc(256 * 256 * 4);
  float* pow1  = (float*)alloc(256 * 256 * 4);
  bf16*  Pb    = (bf16*)alloc(256 * 256 * 2);
  bf16*  Ub    = (bf16*)alloc(256 * 256 * 2);
  bf16*  Wb    = (bf16*)alloc(256 * 256 * 2);
  bf16*  Mb    = (bf16*)alloc(256 * 256 * 2);
  bf16*  states= (bf16*)alloc((size_t)NCH * NDIM * BSZ * 2);

  gj_invert_k<<<1, 1024, 0, stream>>>(A, G);
  abar_k<<<256, 256, 0, stream>>>(G, A, Abar);
  bbar_k<<<1, 256, 0, stream>>>(G, Bv, Bbar);
  chains_k<<<1, 256, 0, stream>>>(Abar, Bbar, Cv, Kk, Uf, Wf);

  const float* src = Abar;
  float* dst = pow0;
  for (int i = 0; i < 8; ++i) {
    square_k<<<256, 256, 0, stream>>>(src, dst);
    src = dst;
    dst = (src == pow0) ? pow1 : pow0;
  }
  tables_k<<<dim3(256, 4), 256, 0, stream>>>(src, Uf, Wf, Kk, Pb, Ub, Wb, Mb);

  size_t smem1 = (size_t)(NDIM * NDIM * 2 + 2 * 8 * 512) * sizeof(bf16);
  phase1_k<<<32, 256, smem1, stream>>>(x, Pb, Ub, states);

  size_t smem2 = (size_t)(NDIM * NDIM * 2 + 2 * 16 * 512) * sizeof(bf16);
  phase2_k<<<dim3(NCH, BSZ / 32), 256, smem2, stream>>>(x, states, Mb, Wb, y);
}
